// MiniTFAttention_28424093565153
// MI455X (gfx1250) — matrix operand order repaired and checked by reading, not compiled, not run on hardware
//
#include <hip/hip_runtime.h>
#include <hip/hip_bf16.h>

// ---------- problem constants ----------
#define BB   2
#define TT   2048
#define DD   1024
#define NH   16
#define HH   64
#define D3   (3 * DD)
#define SCALE 0.125f

typedef __attribute__((ext_vector_type(16))) __bf16 v16bf;
typedef __attribute__((ext_vector_type(8)))  float  v8f;
typedef __attribute__((ext_vector_type(4)))  unsigned int u32x4;
typedef __attribute__((ext_vector_type(8)))  int i32x8;
typedef __attribute__((ext_vector_type(4)))  int i32x4;

// GCC-style vector + addrspace types matching the async-LDS builtin signature
typedef int v4i __attribute__((vector_size(16)));
typedef __attribute__((address_space(1))) v4i as1_v4i;
typedef __attribute__((address_space(3))) v4i as3_v4i;

// gfx1250-specific path availability (host pass sees 0 -> fallback parsed)
#if defined(__has_builtin)
#  if __has_builtin(__builtin_amdgcn_global_load_async_to_lds_b128)
#    define HAVE_ASYNC 1
#  endif
#  if __has_builtin(__builtin_amdgcn_tensor_load_to_lds)
#    define HAVE_TDM 1
#  endif
#endif
#ifndef HAVE_ASYNC
#define HAVE_ASYNC 0
#endif
#ifndef HAVE_TDM
#define HAVE_TDM 0
#endif

union FragB16 { unsigned u[8]; v16bf v; };

__device__ __forceinline__ unsigned short f32_to_bf16(float f) {
    unsigned u = __float_as_uint(f);
    unsigned r = u + 0x7FFFu + ((u >> 16) & 1u);   // round-to-nearest-even
    return (unsigned short)(r >> 16);
}

__device__ __forceinline__ v8f wmma_bf16(const v16bf& a, const v16bf& b, const v8f& c) {
    return __builtin_amdgcn_wmma_f32_16x16x32_bf16(
        /*neg_a=*/false, a, /*neg_b=*/false, b,
        /*c_mod=*/(short)0, c, /*reuse_a=*/false, /*reuse_b=*/false);
}

// A-fragment (16x32, 16-bit): lane l in [0,16): M=l; VGPR v holds K pair:
//   k = (v>=4 ? 16 : 0) + (lane>=16 ? 8 : 0) + (v&3)*2 + {0,1}
__device__ __forceinline__ int a_frag_k(int v, int g) {
    return ((v >> 2) << 4) + (g << 3) + ((v & 3) << 1);
}

// ============================================================================
// GEMM: C[M,Nc] = A[M,K] @ W[Nc,K]^T + bias      (block tile 64x128, 8 waves)
// ============================================================================
template <bool A_IS_BF16, bool OUT_IS_BF16>
__global__ __launch_bounds__(256) void gemm_bias_kernel(
    const void* __restrict__ Aw, const float* __restrict__ W,
    const float* __restrict__ bias, void* __restrict__ Cw,
    int M, int Nc, int K)
{
    __shared__ unsigned short sA[64][40];    // 64x32 bf16, padded
    __shared__ unsigned short sW[128][40];   // 128x32 bf16, padded

    const int tid  = threadIdx.x;
    const int w    = tid >> 5;
    const int lane = tid & 31;
    const int g    = lane >> 4;
    const int l    = lane & 15;

    const int rowBase = blockIdx.y * 64;
    const int colBase = blockIdx.x * 128;
    const int wm = (w & 1) * 32;
    const int wn = (w >> 1) * 32;

    v8f acc[2][2] = {};

    for (int kb = 0; kb < K; kb += 32) {
        // prefetch next K-tile while staging this one
        if (kb + 32 < K) {
            if (tid < 64) {
                size_t gi = (size_t)(rowBase + tid) * K + kb + 32;
                __builtin_prefetch(A_IS_BF16
                    ? (const void*)(((const unsigned short*)Aw) + gi)
                    : (const void*)(((const float*)Aw) + gi), 0, 1);
            } else if (tid < 192) {
                __builtin_prefetch(W + (size_t)(colBase + (tid - 64)) * K + kb + 32, 0, 1);
            }
        }
        // -------- stage A tile (64x32) as bf16 --------
        for (int e = tid; e < 64 * 32; e += 256) {
            int r = e >> 5, c = e & 31;
            size_t gi = (size_t)(rowBase + r) * K + kb + c;
            unsigned short h;
            if (A_IS_BF16) h = ((const unsigned short*)Aw)[gi];
            else           h = f32_to_bf16(((const float*)Aw)[gi]);
            sA[r][c] = h;
        }
        // -------- stage W tile (128x32) as bf16 --------
        for (int e = tid; e < 128 * 32; e += 256) {
            int r = e >> 5, c = e & 31;
            sW[r][c] = f32_to_bf16(W[(size_t)(colBase + r) * K + kb + c]);
        }
        __syncthreads();

        FragB16 af[2], bf[2];
        for (int mi = 0; mi < 2; ++mi) {
            int m = wm + mi * 16 + l;
            for (int v = 0; v < 8; ++v)
                af[mi].u[v] = *(const unsigned*)&sA[m][a_frag_k(v, g)];
        }
        for (int ni = 0; ni < 2; ++ni) {
            int n = wn + ni * 16 + l;
            int k0 = g << 4;
            for (int v = 0; v < 8; ++v)
                bf[ni].u[v] = *(const unsigned*)&sW[n][a_frag_k(v, g)];
        }
        for (int mi = 0; mi < 2; ++mi)
            for (int ni = 0; ni < 2; ++ni)
                acc[mi][ni] = wmma_bf16(af[mi].v, bf[ni].v, acc[mi][ni]);
        __syncthreads();
    }

    for (int mi = 0; mi < 2; ++mi)
        for (int ni = 0; ni < 2; ++ni)
            for (int j = 0; j < 8; ++j) {
                int row = rowBase + wm + mi * 16 + (g << 3) + j;
                int col = colBase + wn + ni * 16 + l;
                float vv = acc[mi][ni][j] + bias[col];
                size_t oi = (size_t)row * Nc + col;
                if (OUT_IS_BF16) ((unsigned short*)Cw)[oi] = f32_to_bf16(vv);
                else             ((float*)Cw)[oi] = vv;
            }
}

// ============================================================================
// Attention: one block per (b, head, 32-row q block).
//  K tiles staged by TDM (tensor_load_to_lds), V tiles by async LDS loads,
//  logits + softmax in 270KB dynamic LDS, attn written once, P@V via WMMA.
// ============================================================================
#define LSTRIDE 2052      // logits row stride (floats): 2048 + 4 pad
#define KTSTR   72        // K/V tile row stride (shorts): 64 + 8 pad (16B align)

__global__ __launch_bounds__(256) void attn_kernel(
    const unsigned short* __restrict__ qkv,   // [B,T,3D] bf16 (k|v|q chunks)
    float* __restrict__ attn,                 // [B,N,T,T] fp32
    unsigned short* __restrict__ head_out)    // [B,T,D] bf16
{
    extern __shared__ float smem[];
    float*          logit = smem;                                   // [32][LSTRIDE]
    unsigned short* qt    = (unsigned short*)(smem + 32 * LSTRIDE); // [32][KTSTR]
    unsigned short* kt    = qt + 32 * KTSTR;                        // [64][KTSTR]

    const int tid  = threadIdx.x;
    const int w    = tid >> 5;
    const int lane = tid & 31;
    const int g    = lane >> 4;
    const int l    = lane & 15;

    const int tb = blockIdx.x & 63;
    const int n  = (blockIdx.x >> 6) & 15;
    const int b  = blockIdx.x >> 10;
    const int t0 = tb * 32;

    const unsigned short* base = qkv + (size_t)b * TT * D3;
    const int kOff = n * HH;                 // chunk order: k, v, q
    const int vOff = DD + n * HH;
    const int qOff = 2 * DD + n * HH;

    // ---- stage Q tile (32 x 64) ----
    for (int e = tid; e < 32 * HH; e += 256) {
        int r = e >> 6, c = e & 63;
        qt[r * KTSTR + c] = base[(size_t)(t0 + r) * D3 + qOff + c];
    }
    __syncthreads();

    const int rw = w & 1;                    // 16-row half owned by this wave
    const int sw = w >> 1;                   // 16-col s-subtile within 64-block
    FragB16 aQ[2];
    for (int cs = 0; cs < 2; ++cs) {
        const unsigned short* qrow = qt + (rw * 16 + l) * KTSTR + cs * 32;
        for (int v = 0; v < 8; ++v)
            aQ[cs].u[v] = *(const unsigned*)&qrow[a_frag_k(v, g)];
    }

    // ---- phase 1: logits = scale * Q K^T with causal mask ----
    const unsigned kt_lds = (unsigned)(size_t)(void*)kt;
    for (int sb = 0; sb < TT; sb += 64) {
        const unsigned short* ksrc = base + (size_t)sb * D3 + kOff;
#if HAVE_TDM
        if (w == 0) {
            // D#: 2D tile 64x64 of 2B elems, row stride 3072 elems,
            //     LDS pad 4 DWORDs every 32 DWORDs (-> 72-short rows)
            unsigned long long ga = (unsigned long long)(size_t)ksrc;
            u32x4 g0;
            g0[0] = 1u;                                   // count=1
            g0[1] = kt_lds;                               // lds_addr
            g0[2] = (unsigned)ga;                         // global_addr[31:0]
            g0[3] = (unsigned)((ga >> 32) & 0x1FFFFFFu) | (2u << 30); // ga[56:32]|type=2
            i32x8 g1;
            g1[0] = (1 << 16) | (1 << 20) | (4 << 22) | (3 << 25); // 2B,pad_en,int=32dw,amt=4dw
            g1[1] = (64 << 16);                           // tensor_dim0 = 64
            g1[2] = (64 << 16);                           // tensor_dim1 = 64
            g1[3] = (64 << 16);                           // tile_dim0 = 64
            g1[4] = 64;                                   // tile_dim1 = 64
            g1[5] = D3;                                   // tensor_dim0_stride
            g1[6] = 0;
            g1[7] = 0;
            i32x4 z4 = {0, 0, 0, 0};
#if __clang_major__ >= 23
            i32x8 z8 = {0, 0, 0, 0, 0, 0, 0, 0};
            __builtin_amdgcn_tensor_load_to_lds(g0, g1, z4, z4, z8, 0);
#else
            __builtin_amdgcn_tensor_load_to_lds(g0, g1, z4, z4, 0);
#endif
            __builtin_amdgcn_s_wait_tensorcnt(0);
        }
#else
        for (int u = tid; u < 512; u += 256) {
            int r = u >> 3, seg = u & 7;
            *(uint4*)&kt[r * KTSTR + seg * 8] =
                *(const uint4*)(ksrc + (size_t)r * D3 + seg * 8);
        }
#endif
        __syncthreads();

        const int sBase = sb + sw * 16;
        FragB16 bK[2];
        const unsigned short* krow = kt + (sw * 16 + l) * KTSTR;
        for (int cs = 0; cs < 2; ++cs) {
            int h0 = cs * 32;
            for (int v = 0; v < 8; ++v)
                bK[cs].u[v] = *(const unsigned*)&krow[h0 + a_frag_k(v, g)];
        }
        v8f acc = {};
        acc = wmma_bf16(aQ[0].v, bK[0].v, acc);
        acc = wmma_bf16(aQ[1].v, bK[1].v, acc);

        int m0 = rw * 16 + (g << 3);
        for (int j = 0; j < 8; ++j) {
            int lrow = m0 + j;
            int trow = t0 + lrow;
            int scol = sBase + l;
            float vv = acc[j] * SCALE;
            if (scol > trow) vv = -__builtin_inff();
            logit[lrow * LSTRIDE + scol] = vv;
        }
        __syncthreads();
    }

    // ---- phase 2: softmax (one wave owns 4 rows) + write attn ----
    for (int r = w * 4; r < w * 4 + 4; ++r) {
        float* row = logit + r * LSTRIDE;
        float mx = -__builtin_inff();
        for (int c = lane; c < TT; c += 32) mx = fmaxf(mx, row[c]);
        for (int off = 16; off; off >>= 1) mx = fmaxf(mx, __shfl_xor(mx, off, 32));
        float sum = 0.f;
        for (int c = lane; c < TT; c += 32) {
            float p = __expf(row[c] - mx);
            row[c] = p;
            sum += p;
        }
        for (int off = 16; off; off >>= 1) sum += __shfl_xor(sum, off, 32);
        float inv = 1.0f / sum;
        float* gattn = attn + (((size_t)b * NH + n) * TT + (t0 + r)) * TT;
        for (int c = lane; c < TT; c += 32) {
            float p = row[c] * inv;
            row[c] = p;
            gattn[c] = p;
        }
    }
    __syncthreads();

    // ---- phase 3: O = P @ V  (V tiles staged via async LDS loads) ----
    unsigned short* vt = kt;                 // reuse [32][KTSTR]
    const int cw = w >> 1;                   // 4 col stripes of 16 over H=64
    v8f acc2 = {};
    for (int ss = 0; ss < TT / 32; ++ss) {
        int sb = ss * 32;
        {   // stage V[sb..sb+32)[0..64) -> vt : one b128 per thread
            int r = tid >> 3, seg = tid & 7;
            const unsigned short* s = base + (size_t)(sb + r) * D3 + vOff + seg * 8;
            unsigned short* d = vt + r * KTSTR + seg * 8;
#if HAVE_ASYNC
            __builtin_amdgcn_global_load_async_to_lds_b128(
                (as1_v4i*)s, (as3_v4i*)d, 0, 0);
            asm volatile("s_wait_asynccnt 0x0" ::: "memory");
#else
            *(uint4*)d = *(const uint4*)s;
#endif
        }
        __syncthreads();

        FragB16 aP, bV;
        {
            const float* prow = logit + (rw * 16 + l) * LSTRIDE + sb;
            for (int v = 0; v < 8; ++v) {
                int k = a_frag_k(v, g);
                unsigned short lo = f32_to_bf16(prow[k]);
                unsigned short hi = f32_to_bf16(prow[k + 1]);
                aP.u[v] = (unsigned)lo | ((unsigned)hi << 16);
            }
        }
        {
            int c = cw * 16 + l;
            for (int v = 0; v < 8; ++v) {
                int k0 = a_frag_k(v, g);
                unsigned short lo = vt[k0 * KTSTR + c];
                unsigned short hi = vt[(k0 + 1) * KTSTR + c];
                bV.u[v] = (unsigned)lo | ((unsigned)hi << 16);
            }
        }
        acc2 = wmma_bf16(aP.v, bV.v, acc2);
        __syncthreads();
    }
    for (int j = 0; j < 8; ++j) {
        int trow = t0 + rw * 16 + (g << 3) + j;
        int col  = n * HH + cw * 16 + l;
        head_out[((size_t)b * TT + trow) * DD + col] = f32_to_bf16(acc2[j]);
    }
}

// ============================================================================
extern "C" void kernel_launch(void* const* d_in, const int* in_sizes, int n_in,
                              void* d_out, int out_size, void* d_ws, size_t ws_size,
                              hipStream_t stream) {
    const float* x  = (const float*)d_in[0];   // [2,2048,1024]
    const float* pw = (const float*)d_in[1];   // [3072,1024]
    const float* pb = (const float*)d_in[2];   // [3072]
    const float* ow = (const float*)d_in[3];   // [1024,1024]
    const float* ob = (const float*)d_in[4];   // [1024]
    // d_in[5] = causal mask (recomputed analytically in-kernel)

    float* out  = (float*)d_out;                              // [2,2048,1024]
    float* attn = out + (size_t)BB * TT * DD;                 // [2,16,2048,2048]

    unsigned short* qkv  = (unsigned short*)d_ws;             // [2,2048,3072] bf16
    unsigned short* hout = qkv + (size_t)BB * TT * D3;        // [2,2048,1024] bf16

    const int M = BB * TT;   // 4096
    dim3 blk(256);

    // 1) qkv projection (fp32 in -> bf16 out)
    gemm_bias_kernel<false, true>
        <<<dim3(D3 / 128, M / 64), blk, 0, stream>>>(x, pw, pb, qkv, M, D3, DD);

    // 2) attention (TDM K-staging, logits, softmax, attn write, async-V P@V)
    size_t lds = (size_t)32 * LSTRIDE * sizeof(float)
               + (size_t)32 * KTSTR * sizeof(unsigned short)
               + (size_t)64 * KTSTR * sizeof(unsigned short);
    attn_kernel<<<dim3(BB * NH * (TT / 32)), blk, lds, stream>>>(qkv, attn, hout);

    // 3) output projection (bf16 in -> fp32 out)
    gemm_bias_kernel<true, false>
        <<<dim3(DD / 128, M / 64), blk, 0, stream>>>(hout, ow, ob, out, M, DD, DD);
}
